// AttentionHelper_88880053224156
// MI455X (gfx1250) — matrix operand order repaired and checked by reading, not compiled, not run on hardware
//
#include <hip/hip_runtime.h>
#include <math.h>

typedef __attribute__((ext_vector_type(16))) _Float16 v16h;
typedef __attribute__((ext_vector_type(8)))  float    v8f;
typedef __attribute__((ext_vector_type(4)))  float    v4f;

#define B_  8
#define C_  128
#define LQ_ 2048
#define LK_ 2048

// log(1e-9f) in f32; log(1.0f + 1e-9f) == 0.0f exactly in f32.
#define LOG_EPS (-20.723265865f)

// Fused masked-attention: one block per (batch, 16-query tile).
// 256 threads = 8 wave32 waves. Dynamic LDS ~140 KB (CDNA5: 320KB/WGP).
__global__ __launch_bounds__(256) void attn_fused_kernel(
    const float* __restrict__ Q, const float* __restrict__ Kg,
    const float* __restrict__ V, const int* __restrict__ mask,
    float* __restrict__ out, float* __restrict__ att)
{
    extern __shared__ char smem[];
    float* s_att = (float*)smem;                      // [16][2048] energy -> exp(p)
    float* s_q   = (float*)(smem + 16 * LK_ * 4);     // [128][16]
    float* s_red = s_q + C_ * 16;                     // [16][16] reduction scratch
    float* s_inv = s_red + 256;                       // [16] 1/rowsum
    float* s_max = s_inv + 16;                        // [16] rowmax

    const int b    = blockIdx.x / (LQ_ / 16);
    const int q0   = (blockIdx.x % (LQ_ / 16)) * 16;
    const int tid  = threadIdx.x;
    const int wave = tid >> 5;
    const int lane = tid & 31;
    const int lm   = lane & 15;          // N / M-in-halfwave index
    const int lh   = lane >> 4;          // half-wave select
    const float scale = 0.08838834764831845f; // 1/sqrt(128)

    const int* Mb = mask + (size_t)(b * LQ_ + q0) * LK_;

    // prefetch the mask tile (16 x 2048 ints = 128KB) while we compute energy
    for (int i = tid; i < 1024; i += 256)
        __builtin_prefetch(Mb + i * 32, 0, 1);

    // ---- stage Q tile (C x 16) into LDS with b128 loads ----
    {
        const float* Qb = Q + (size_t)b * C_ * LQ_ + q0;
        for (int i = tid; i < C_ * 16 / 4; i += 256) {
            int c = i >> 2, q4 = (i & 3) * 4;
            *(v4f*)&s_q[c * 16 + q4] = *(const v4f*)&Qb[(size_t)c * LQ_ + q4];
        }
    }
    __syncthreads();

    // ---- build 4 Q^T A-fragments (16x32 f16), ISA 7.12.2 layout ----
    // lane<16: M=lm, K = {0..7,16..23}; lane>=16: M=lm, K = {8..15,24..31}
    v16h aq[4];
#pragma unroll
    for (int cc = 0; cc < 4; ++cc) {
#pragma unroll
        for (int e = 0; e < 8; ++e) {
            int cA = cc * 32 + lh * 8 + e;
            int cB = cc * 32 + 16 + lh * 8 + e;
            aq[cc][e]     = (_Float16)s_q[cA * 16 + lm];
            aq[cc][e + 8] = (_Float16)s_q[cB * 16 + lm];
        }
    }

    // ---- phase 1: energy = scale*(Q^T K) + log(mask+1e-9) into LDS ----
    // wave w covers k in [w*256, w*256+256): 16 subtiles of 16 columns
    {
        const float* Kb = Kg + (size_t)b * C_ * LK_;
        for (int ns = 0; ns < 16; ++ns) {
            const int kb = wave * 256 + ns * 16;
            v8f acc = {};
#pragma unroll
            for (int cc = 0; cc < 4; ++cc) {
                // K B-fragment (32x16): lane col N=lm, rows c = cc*32 + lh*8 + e / cc*32 + 16 + lh*8 + (e-8)
                v16h bk;
#pragma unroll
                for (int e = 0; e < 16; ++e) {
                    int c = cc * 32 + ((e < 8) ? (lh * 8 + e) : (16 + lh * 8 + (e - 8)));
                    bk[e] = (_Float16)Kb[(size_t)c * LK_ + kb + lm];
                }
                acc = __builtin_amdgcn_wmma_f32_16x16x32_f16(
                    false, aq[cc], false, bk, (short)0, acc, false, false);
            }
            // epilogue: D layout -> lanes0-15 M=r, lanes16-31 M=8+r; N=lm
#pragma unroll
            for (int r = 0; r < 8; ++r) {
                int row = (lane < 16) ? r : (8 + r);
                int k   = kb + lm;
                // log(mask + 1e-9) == (mask ? 0.0f : LOG_EPS) exactly in f32
                float lterm = (Mb[(size_t)row * LK_ + k] != 0) ? 0.0f : LOG_EPS;
                s_att[row * LK_ + k] = acc[r] * scale + lterm;
            }
        }
    }
    __syncthreads();

    // ---- softmax over each of the 16 rows (2048 wide) in LDS, b128 sweeps ----
    {
        const int r = tid & 15;    // row
        const int t = tid >> 4;    // team member 0..15, 128-elem segment
        float* rowp = s_att + r * LK_;
        v4f mx4 = { -3.0e38f, -3.0e38f, -3.0e38f, -3.0e38f };
        for (int k = t * 128; k < t * 128 + 128; k += 4) {
            v4f v = *(v4f*)&rowp[k];
#pragma unroll
            for (int j = 0; j < 4; ++j) mx4[j] = fmaxf(mx4[j], v[j]);
        }
        float mx = fmaxf(fmaxf(mx4[0], mx4[1]), fmaxf(mx4[2], mx4[3]));
        s_red[r * 16 + t] = mx;
        __syncthreads();
        if (tid < 16) {
            float v = s_red[tid * 16];
            for (int j = 1; j < 16; ++j) v = fmaxf(v, s_red[tid * 16 + j]);
            s_max[tid] = v;
        }
        __syncthreads();
        float rmax = s_max[r];
        float sum = 0.0f;
        for (int k = t * 128; k < t * 128 + 128; k += 4) {
            v4f v = *(v4f*)&rowp[k];
            v4f p;
#pragma unroll
            for (int j = 0; j < 4; ++j) p[j] = __expf(v[j] - rmax);
            *(v4f*)&rowp[k] = p;
            sum += (p[0] + p[1]) + (p[2] + p[3]);
        }
        s_red[r * 16 + t] = sum;
        __syncthreads();
        if (tid < 16) {
            float v = 0.0f;
            for (int j = 0; j < 16; ++j) v += s_red[tid * 16 + j];
            s_inv[tid] = 1.0f / v;
        }
        __syncthreads();
    }

    // ---- write normalized attention: b128 LDS reads + b128 global stores ----
    {
        float* attb = att + (size_t)(b * LQ_ + q0) * LK_;
        for (int i = tid; i < 16 * LK_ / 4; i += 256) {
            int idx = i * 4;
            int row = idx >> 11;           // /2048
            v4f v = *(v4f*)&s_att[idx];
            v = v * s_inv[row];
            *(v4f*)&attb[idx] = v;
        }
    }

    // ---- phase 2: out = V * A^T ; wave w owns c-subtile [w*16, w*16+16) ----
    {
        const float* Vb = V + (size_t)b * C_ * LK_;
        const int c0 = wave * 16;
        v8f acc = {};
        for (int kc = 0; kc < LK_; kc += 32) {
            // V A-fragment: M=lm -> c=c0+lm; K = kc + lh*8 + e  /  kc+16+lh*8+e
            const float* vrow = Vb + (size_t)(c0 + lm) * LK_ + kc + lh * 8;
            v4f v0 = *(const v4f*)&vrow[0];
            v4f v1 = *(const v4f*)&vrow[4];
            v4f v2 = *(const v4f*)&vrow[16];
            v4f v3 = *(const v4f*)&vrow[20];
            v16h av;
#pragma unroll
            for (int j = 0; j < 4; ++j) {
                av[j]      = (_Float16)v0[j];
                av[4 + j]  = (_Float16)v1[j];
                av[8 + j]  = (_Float16)v2[j];
                av[12 + j] = (_Float16)v3[j];
            }
            // attention B-fragment from LDS: N=lm (query row), K = kc + lh*8 + e  /  kc+16+lh*8+e
            const float* arow = s_att + lm * LK_ + kc + lh * 8;
            v4f a0 = *(const v4f*)&arow[0];
            v4f a1 = *(const v4f*)&arow[4];
            v4f a2 = *(const v4f*)&arow[16];
            v4f a3 = *(const v4f*)&arow[20];
            v16h ba;
#pragma unroll
            for (int j = 0; j < 4; ++j) {
                ba[j]      = (_Float16)a0[j];
                ba[4 + j]  = (_Float16)a1[j];
                ba[8 + j]  = (_Float16)a2[j];
                ba[12 + j] = (_Float16)a3[j];
            }
            acc = __builtin_amdgcn_wmma_f32_16x16x32_f16(
                false, av, false, ba, (short)0, acc, false, false);
        }
        // epilogue: out[b][c][q0+N] = acc[r] * invsum[N]
        float* outb = out + (size_t)(b * C_) * LQ_ + q0;
        float inv = s_inv[lm];
#pragma unroll
        for (int r = 0; r < 8; ++r) {
            int c = c0 + ((lane < 16) ? r : (8 + r));
            outb[(size_t)c * LQ_ + lm] = acc[r] * inv;
        }
    }
}

extern "C" void kernel_launch(void* const* d_in, const int* in_sizes, int n_in,
                              void* d_out, int out_size, void* d_ws, size_t ws_size,
                              hipStream_t stream) {
    (void)in_sizes; (void)n_in; (void)out_size; (void)d_ws; (void)ws_size;
    const float* Q    = (const float*)d_in[0];
    const float* Kg   = (const float*)d_in[1];
    const float* V    = (const float*)d_in[2];
    const int*   mask = (const int*)d_in[3];
    float* out = (float*)d_out;                          // (B, C, LQ)
    float* att = out + (size_t)B_ * C_ * LQ_;            // (B, LQ, LK)

    const size_t lds_bytes = (size_t)16 * LK_ * 4   // s_att
                           + (size_t)C_ * 16 * 4    // s_q
                           + 256 * 4 + 16 * 4 + 16 * 4;

    // raise dynamic-LDS cap (CDNA5 WGP has 320KB); host-side, capture-safe
    (void)hipFuncSetAttribute((const void*)attn_fused_kernel,
                              hipFuncAttributeMaxDynamicSharedMemorySize,
                              (int)lds_bytes);

    dim3 grid(B_ * (LQ_ / 16));
    attn_fused_kernel<<<grid, 256, lds_bytes, stream>>>(Q, Kg, V, mask, out, att);
}
